// Attention_47141561041147
// MI455X (gfx1250) — matrix operand order repaired and checked by reading, not compiled, not run on hardware
//
#include <hip/hip_runtime.h>
#include <hip/hip_bf16.h>

// ---------------------------------------------------------------------------
// Attention layer for MI455X (gfx1250, wave32, WMMA + Tensor Data Mover).
//   q,k,v = x @ w{q,k,v}^T ; RoPE(q,k) ; softmax(q k^T / sqrt(hd)) v ; @ wo^T
// Matmuls on v_wmma_f32_16x16x32_bf16; LDS tiles staged with tensor_load_to_lds
// (TDM, TENSORcnt), double-buffered to overlap DMA with WMMA.
// B=2, S=2048, D=2048, H=16, hd=128. Workspace: ~112 MB.
// ---------------------------------------------------------------------------

#define B_SZ   2
#define S_LEN  2048
#define D_DIM  2048
#define HEADS  16
#define HDIM   128
#define BS     (B_SZ * S_LEN)       // 4096

typedef __attribute__((ext_vector_type(16))) __bf16 v16bf;
typedef __attribute__((ext_vector_type(8)))  __bf16 v8bf;
typedef __attribute__((ext_vector_type(4)))  __bf16 v4bf;
typedef __attribute__((ext_vector_type(8)))  float  v8f;
typedef __attribute__((ext_vector_type(4)))  unsigned int v4u;
typedef __attribute__((ext_vector_type(8)))  int v8i;
typedef __attribute__((ext_vector_type(4)))  int v4i;

__device__ inline v16bf frag_cat(v8bf lo, v8bf hi) {
  v16bf r;
#pragma unroll
  for (int i = 0; i < 8; ++i) { r[i] = lo[i]; r[i + 8] = hi[i]; }
  return r;
}

// A fragment: 16x32 bf16 tile, row-major source with row stride `ld` elements.
// Lanes 0-15: row=lane, K 0..15; lanes 16-31: row=lane-16, K 16..31.
__device__ inline v16bf load_afrag(const __bf16* p, int ld) {
  int lane = threadIdx.x & 31;
  int m = lane & 15, h = lane >> 4;
  const __bf16* r = p + (size_t)m * ld + h * 16;
  return frag_cat(*(const v8bf*)r, *(const v8bf*)(r + 8));
}

// B fragment for "X @ W^T": B[k][n] = W[n][k], W row-major, stride ld.
// Lanes 0-15: col n=lane, K 0..15; lanes 16-31: n=lane-16, K 16..31.
__device__ inline v16bf load_bfrag_rowmajor(const __bf16* w, int ld) {
  int lane = threadIdx.x & 31;
  int n = lane & 15, h = lane >> 4;
  const __bf16* r = w + (size_t)n * ld + h * 16;
  return frag_cat(*(const v8bf*)r, *(const v8bf*)(r + 8));
}

__device__ inline v8f wmma_bf16(v16bf a, v16bf b, v8f c) {
  return __builtin_amdgcn_wmma_f32_16x16x32_bf16(false, a, false, b,
                                                 (short)0, c, false, false);
}

// ---------------------------------------------------------------------------
// TDM: DMA a 2-D tile of bf16 (tile_d1 rows x tile_d0 elements, row stride
// `stride` elements in memory) into dense row-major LDS at lds_off.
// D# layout per CDNA5 ISA ch.8 (group0: count/lds/global/type; group1:
// data_size, tensor dims, tile dims, dim0 stride). Groups 2/3 zero (<=2-D).
// 6-arg builtin variant (clang-23 / therock headers).
// ---------------------------------------------------------------------------
__device__ inline void tdm_load_2d(unsigned lds_off, const __bf16* gsrc,
                                   unsigned tile_d0, unsigned tile_d1,
                                   unsigned stride,
                                   unsigned tensor_d0, unsigned tensor_d1) {
  unsigned long long ga = (unsigned long long)(size_t)gsrc;
  v4u g0;
  g0[0] = 1u;                                       // count=1, user D#
  g0[1] = lds_off;                                  // lds_addr (bytes)
  g0[2] = (unsigned)(ga & 0xffffffffu);             // global_addr[31:0]
  g0[3] = (unsigned)((ga >> 32) & 0x01ffffffu)      // global_addr[56:32]
          | 0x80000000u;                            // type=2 ("image")
  v8i g1;
  g1[0] = (int)(1u << 16);                          // data_size=1 -> 2 bytes
  g1[1] = (int)((tensor_d0 & 0xffffu) << 16);       // tensor_dim0[15:0]
  g1[2] = (int)(((tensor_d0 >> 16) & 0xffffu)       // tensor_dim0[31:16]
          | ((tensor_d1 & 0xffffu) << 16));         // tensor_dim1[15:0]
  g1[3] = (int)(((tensor_d1 >> 16) & 0xffffu)       // tensor_dim1[31:16]
          | ((tile_d0 & 0xffffu) << 16));           // tile_dim0
  g1[4] = (int)(tile_d1 & 0xffffu);                 // tile_dim1 (tile_dim2=0)
  g1[5] = (int)stride;                              // tensor_dim0_stride[31:0]
  g1[6] = 0;                                        // stride hi / dim1_stride
  g1[7] = 0;
  v4i z4 = {};
  v8i z8 = {};
  __builtin_amdgcn_tensor_load_to_lds(g0, g1, z4, z4, z8, 0);
}

__device__ inline unsigned lds_offset_of(const void* p) {
  return (unsigned)(unsigned long long)(size_t)p;    // low 32 bits = LDS offset
}

// ---------------------------------------------------------------------------
// fp32 -> bf16 cast
// ---------------------------------------------------------------------------
__global__ __launch_bounds__(256) void cast_f32_bf16(
    const float* __restrict__ in, __bf16* __restrict__ out, int n) {
  int i = (blockIdx.x * 256 + threadIdx.x) * 4;
  if (i + 3 < n) {
    float4 f = *(const float4*)(in + i);
    v4bf o;
    o[0] = (__bf16)f.x; o[1] = (__bf16)f.y;
    o[2] = (__bf16)f.z; o[3] = (__bf16)f.w;
    *(v4bf*)(out + i) = o;
  }
}

// ---------------------------------------------------------------------------
// Fused QKV GEMM + RoPE epilogue. K-step 64, TDM-staged A tile, double buffer.
// grid = (BS/16, D/512, 3 [q,k,v]); block = 256 (8 waves); wave -> 16x64 strip.
// Q,K written as [b][h][s][d]; V written transposed as [b][h][d][s].
// ---------------------------------------------------------------------------
__global__ __launch_bounds__(256) void qkv_gemm_rope(
    const __bf16* __restrict__ X,
    const __bf16* __restrict__ Wq, const __bf16* __restrict__ Wk,
    const __bf16* __restrict__ Wv,
    __bf16* __restrict__ Qo, __bf16* __restrict__ Ko,
    __bf16* __restrict__ Vt) {
  __shared__ __attribute__((aligned(16))) __bf16 xs[2][16 * 64];

  const int z = blockIdx.z;
  const __bf16* W = (z == 0) ? Wq : (z == 1) ? Wk : Wv;
  const int m0 = blockIdx.x * 16;
  const int wave = threadIdx.x >> 5;
  const int lane = threadIdx.x & 31;
  const int n0 = blockIdx.y * 512 + wave * 64;
  const bool w0 = (threadIdx.x < 32);   // wave-uniform: wave 0 drives the TDM
  const __bf16* xrow = X + (size_t)m0 * D_DIM;

  v8f vzero = {};
  v8f acc[4] = {vzero, vzero, vzero, vzero};

  if (w0)
    tdm_load_2d(lds_offset_of(&xs[0][0]), xrow, 64, 16, D_DIM, D_DIM, BS);

  int cur = 0;
  for (int k0 = 0; k0 < D_DIM; k0 += 64) {
    if (w0) __builtin_amdgcn_s_wait_tensorcnt((short)0);
    __syncthreads();                       // tile[cur] ready; tile[cur^1] free
    if (w0 && (k0 + 64 < D_DIM))
      tdm_load_2d(lds_offset_of(&xs[cur ^ 1][0]), xrow + k0 + 64,
                  64, 16, D_DIM, D_DIM, BS);

    v16bf a0 = load_afrag(&xs[cur][0], 64);
    v16bf a1 = load_afrag(&xs[cur][32], 64);
#pragma unroll
    for (int t = 0; t < 4; ++t) {
      const __bf16* wp = W + (size_t)(n0 + t * 16) * D_DIM + k0;
      __builtin_prefetch(wp + 64, 0, 0);   // global_prefetch for next K-step
      v16bf b0 = load_bfrag_rowmajor(wp, D_DIM);
      v16bf b1 = load_bfrag_rowmajor(wp + 32, D_DIM);
      acc[t] = wmma_bf16(a0, b0, acc[t]);
      acc[t] = wmma_bf16(a1, b1, acc[t]);
    }
    cur ^= 1;
  }
  __syncthreads();

  const int nq = lane & 15;
  const int mbase = (lane >> 4) * 8;

  if (z < 2) {
    // RoPE + store [b][h][s][d]. Pairs (2j,2j+1) live in adjacent lanes.
    __bf16* O = (z == 0) ? Qo : Ko;
#pragma unroll
    for (int t = 0; t < 4; ++t) {
      int gn = n0 + t * 16 + nq;
      int h = gn >> 7, d = gn & 127;
      int j2 = d & ~1;  // 2j
      float inv = __powf(10000.0f, -(float)j2 / 128.0f);
#pragma unroll
      for (int i = 0; i < 8; ++i) {
        int gm = m0 + mbase + i;
        int bidx = gm >> 11;
        int s = gm & (S_LEN - 1);
        float ang = (float)s * inv;
        float cs = __cosf(ang), sn = __sinf(ang);
        float val = acc[t][i];
        float other = __shfl_xor(val, 1, 32);
        float res = ((d & 1) == 0) ? (val * cs - other * sn)
                                   : (other * sn + val * cs);
        O[((size_t)(bidx * HEADS + h) * S_LEN + s) * HDIM + d] = (__bf16)res;
      }
    }
  } else {
    // V transposed store: [b][h][d][s]
#pragma unroll
    for (int t = 0; t < 4; ++t) {
      int gn = n0 + t * 16 + nq;
      int h = gn >> 7, d = gn & 127;
#pragma unroll
      for (int i = 0; i < 8; ++i) {
        int gm = m0 + mbase + i;
        int bidx = gm >> 11;
        int s = gm & (S_LEN - 1);
        Vt[((size_t)(bidx * HEADS + h) * HDIM + d) * S_LEN + s] = (__bf16)acc[t][i];
      }
    }
  }
}

// ---------------------------------------------------------------------------
// Flash attention. 4 waves/block share one (b,h); K and V tiles are TDM-staged
// into LDS (double-buffered) and consumed by all 4 waves; online softmax over
// 2048 keys in steps of 32. grid = (S/64, H, B); block = 128.
// ---------------------------------------------------------------------------
__global__ __launch_bounds__(128) void flash_attn(
    const __bf16* __restrict__ Q, const __bf16* __restrict__ K,
    const __bf16* __restrict__ Vt, __bf16* __restrict__ Ao) {
  __shared__ __attribute__((aligned(16))) __bf16 kbuf[2][32 * HDIM];  // 32 keys x 128
  __shared__ __attribute__((aligned(16))) __bf16 vbuf[2][HDIM * 32];  // 128 d x 32 s
  __shared__ __attribute__((aligned(16))) __bf16 pbuf[4][16 * 32];

  const int wave = threadIdx.x >> 5;
  const int lane = threadIdx.x & 31;
  const int qt = blockIdx.x * 4 + wave;  // 16-row query tile
  const int h = blockIdx.y;
  const int b = blockIdx.z;
  const size_t bh = (size_t)(b * HEADS + h);
  const int nq = lane & 15;
  const int mbase = (lane >> 4) * 8;
  const bool w0 = (threadIdx.x < 32);
  const float scale = 0.08838834764831845f;  // 1/sqrt(128)

  const __bf16* kbase = K + bh * S_LEN * HDIM;
  const __bf16* vbase = Vt + bh * HDIM * S_LEN;

  // Resident Q fragments: 16x128 = 4 K-chunks of 32
  const __bf16* qp = Q + (bh * S_LEN + (size_t)qt * 16) * HDIM;
  v16bf qa[4];
#pragma unroll
  for (int c = 0; c < 4; ++c) qa[c] = load_afrag(qp + c * 32, HDIM);

  v8f vzero = {};
  float mi[8], li[8];
  v8f o[8];
#pragma unroll
  for (int i = 0; i < 8; ++i) { mi[i] = -1e30f; li[i] = 0.0f; }
#pragma unroll
  for (int t = 0; t < 8; ++t) o[t] = vzero;

  __bf16* plds = pbuf[wave];

  if (w0) {
    // K tile: 32 rows x 128 elems, stride 128. V tile: 128 rows x 32, stride S.
    tdm_load_2d(lds_offset_of(&kbuf[0][0]), kbase, HDIM, 32, HDIM, HDIM, S_LEN);
    tdm_load_2d(lds_offset_of(&vbuf[0][0]), vbase, 32, HDIM, S_LEN, S_LEN, HDIM);
  }

  int cur = 0;
  for (int j0 = 0; j0 < S_LEN; j0 += 32) {
    if (w0) __builtin_amdgcn_s_wait_tensorcnt((short)0);
    __syncthreads();                     // tiles[cur] ready; tiles[cur^1] free
    if (w0 && (j0 + 32 < S_LEN)) {
      tdm_load_2d(lds_offset_of(&kbuf[cur ^ 1][0]), kbase + (size_t)(j0 + 32) * HDIM,
                  HDIM, 32, HDIM, HDIM, S_LEN);
      tdm_load_2d(lds_offset_of(&vbuf[cur ^ 1][0]), vbase + j0 + 32,
                  32, HDIM, S_LEN, S_LEN, HDIM);
    }

    // scores S = Q K^T for 32 keys -> two 16x16 fp32 tiles (from LDS K tile)
    v8f s0 = vzero, s1 = vzero;
    const __bf16* kp = &kbuf[cur][0];
#pragma unroll
    for (int c = 0; c < 4; ++c) {
      v16bf b0 = load_bfrag_rowmajor(kp + c * 32, HDIM);
      v16bf b1 = load_bfrag_rowmajor(kp + 16 * HDIM + c * 32, HDIM);
      s0 = wmma_bf16(qa[c], b0, s0);
      s1 = wmma_bf16(qa[c], b1, s1);
    }

    // online softmax per row (rows in VGPR index, cols across a 16-lane half)
#pragma unroll
    for (int i = 0; i < 8; ++i) {
      float a0 = s0[i] * scale, a1 = s1[i] * scale;
      float rm = fmaxf(a0, a1);
      rm = fmaxf(rm, __shfl_xor(rm, 1, 32));
      rm = fmaxf(rm, __shfl_xor(rm, 2, 32));
      rm = fmaxf(rm, __shfl_xor(rm, 4, 32));
      rm = fmaxf(rm, __shfl_xor(rm, 8, 32));
      float mnew = fmaxf(mi[i], rm);
      float alpha = __expf(mi[i] - mnew);
      float p0 = __expf(a0 - mnew);
      float p1 = __expf(a1 - mnew);
      float ps = p0 + p1;
      ps += __shfl_xor(ps, 1, 32);
      ps += __shfl_xor(ps, 2, 32);
      ps += __shfl_xor(ps, 4, 32);
      ps += __shfl_xor(ps, 8, 32);
      li[i] = li[i] * alpha + ps;
      mi[i] = mnew;
#pragma unroll
      for (int t = 0; t < 8; ++t) o[t][i] = o[t][i] * alpha;
      // P: D-layout -> row-major 16x32 in LDS (A-layout bounce, wave-private)
      plds[(mbase + i) * 32 + nq]      = (__bf16)p0;
      plds[(mbase + i) * 32 + 16 + nq] = (__bf16)p1;
    }

    v16bf pa = load_afrag(plds, 32);  // 16x32 probability fragment
    // O += P V from LDS V tile: B[k][n] = vbuf[t*16+n][k]
#pragma unroll
    for (int t = 0; t < 8; ++t) {
      const __bf16* vr = &vbuf[cur][(t * 16 + nq) * 32 + (lane >> 4) * 16];
      v16bf vb = frag_cat(*(const v8bf*)vr, *(const v8bf*)(vr + 8));
      o[t] = wmma_bf16(pa, vb, o[t]);
    }
    __syncthreads();                   // all waves done with tiles[cur]
    cur ^= 1;
  }

  // normalize and store row-major [b*S+s][D] for the output GEMM
#pragma unroll
  for (int t = 0; t < 8; ++t) {
#pragma unroll
    for (int i = 0; i < 8; ++i) {
      float inv_l = 1.0f / li[i];
      Ao[((size_t)(b * S_LEN) + qt * 16 + mbase + i) * D_DIM + h * HDIM + t * 16 + nq] =
          (__bf16)(o[t][i] * inv_l);
    }
  }
}

// ---------------------------------------------------------------------------
// Output GEMM: out = attn_bf16 @ wo^T -> fp32. Same TDM tiling as QKV GEMM.
// ---------------------------------------------------------------------------
__global__ __launch_bounds__(256) void out_gemm(
    const __bf16* __restrict__ A, const __bf16* __restrict__ Wo,
    float* __restrict__ Out) {
  __shared__ __attribute__((aligned(16))) __bf16 as[2][16 * 64];

  const int m0 = blockIdx.x * 16;
  const int wave = threadIdx.x >> 5;
  const int lane = threadIdx.x & 31;
  const int n0 = blockIdx.y * 512 + wave * 64;
  const bool w0 = (threadIdx.x < 32);
  const __bf16* arow = A + (size_t)m0 * D_DIM;

  v8f vzero = {};
  v8f acc[4] = {vzero, vzero, vzero, vzero};

  if (w0)
    tdm_load_2d(lds_offset_of(&as[0][0]), arow, 64, 16, D_DIM, D_DIM, BS);

  int cur = 0;
  for (int k0 = 0; k0 < D_DIM; k0 += 64) {
    if (w0) __builtin_amdgcn_s_wait_tensorcnt((short)0);
    __syncthreads();
    if (w0 && (k0 + 64 < D_DIM))
      tdm_load_2d(lds_offset_of(&as[cur ^ 1][0]), arow + k0 + 64,
                  64, 16, D_DIM, D_DIM, BS);

    v16bf a0 = load_afrag(&as[cur][0], 64);
    v16bf a1 = load_afrag(&as[cur][32], 64);
#pragma unroll
    for (int t = 0; t < 4; ++t) {
      const __bf16* wp = Wo + (size_t)(n0 + t * 16) * D_DIM + k0;
      __builtin_prefetch(wp + 64, 0, 0);
      v16bf b0 = load_bfrag_rowmajor(wp, D_DIM);
      v16bf b1 = load_bfrag_rowmajor(wp + 32, D_DIM);
      acc[t] = wmma_bf16(a0, b0, acc[t]);
      acc[t] = wmma_bf16(a1, b1, acc[t]);
    }
    cur ^= 1;
  }
  __syncthreads();

  const int nq = lane & 15;
  const int mbase = (lane >> 4) * 8;
#pragma unroll
  for (int t = 0; t < 4; ++t) {
#pragma unroll
    for (int i = 0; i < 8; ++i) {
      Out[(size_t)(m0 + mbase + i) * D_DIM + n0 + t * 16 + nq] = acc[t][i];
    }
  }
}

// ---------------------------------------------------------------------------
extern "C" void kernel_launch(void* const* d_in, const int* in_sizes, int n_in,
                              void* d_out, int out_size, void* d_ws, size_t ws_size,
                              hipStream_t stream) {
  (void)in_sizes; (void)n_in; (void)out_size; (void)ws_size;
  const float* x  = (const float*)d_in[0];
  const float* wq = (const float*)d_in[1];
  const float* wk = (const float*)d_in[2];
  const float* wv = (const float*)d_in[3];
  const float* wo = (const float*)d_in[4];

  const int nX = BS * D_DIM;        // 8388608
  const int nW = D_DIM * D_DIM;     // 4194304
  const int nQ = BS * D_DIM;        // per q/k/vT/attn buffer

  __bf16* xb  = (__bf16*)d_ws;
  __bf16* wqb = xb  + nX;
  __bf16* wkb = wqb + nW;
  __bf16* wvb = wkb + nW;
  __bf16* wob = wvb + nW;
  __bf16* qb  = wob + nW;
  __bf16* kb  = qb  + nQ;
  __bf16* vtb = kb  + nQ;
  __bf16* ab  = vtb + nQ;   // total ~112 MB of workspace

  cast_f32_bf16<<<nX / 4 / 256, 256, 0, stream>>>(x,  xb,  nX);
  cast_f32_bf16<<<nW / 4 / 256, 256, 0, stream>>>(wq, wqb, nW);
  cast_f32_bf16<<<nW / 4 / 256, 256, 0, stream>>>(wk, wkb, nW);
  cast_f32_bf16<<<nW / 4 / 256, 256, 0, stream>>>(wv, wvb, nW);
  cast_f32_bf16<<<nW / 4 / 256, 256, 0, stream>>>(wo, wob, nW);

  qkv_gemm_rope<<<dim3(BS / 16, D_DIM / 512, 3), 256, 0, stream>>>(
      xb, wqb, wkb, wvb, qb, kb, vtb);

  flash_attn<<<dim3(S_LEN / 64, HEADS, B_SZ), 128, 0, stream>>>(qb, kb, vtb, ab);

  out_gemm<<<dim3(BS / 16, D_DIM / 512), 256, 0, stream>>>(ab, wob, (float*)d_out);
}
